// FeatureTransformer_45896020525219
// MI455X (gfx1250) — matrix operand order repaired and checked by reading, not compiled, not run on hardware
//
#include <hip/hip_runtime.h>
#include <hip/hip_bf16.h>

typedef __attribute__((ext_vector_type(16))) _Float16 v16h;
typedef __attribute__((ext_vector_type(8)))  float    v8f;

#define K_FEAT 32
#define NOUT   256
#define WAVES_PER_BLOCK 8
#define THREADS (WAVES_PER_BLOCK * 32)

// One wave per batch row.
// out[b, 16t+n] = bias[16t+n] + sum_k vals[b,k] * W[ics[b,k], 16t+n]
// computed as one v_wmma_f32_16x16x32_f16 per 16-column tile:
//   A(16x32)  = vals row, replicated across all 16 M rows (f16)
//   B(32x16)  = gathered 16-col slices of the 32 weight rows (f16)
//   D(16x16)  = out tile replicated across M; f32 accumulate
__global__ __launch_bounds__(THREADS)
void FeatureTransformer_embbag_wmma(const int*   __restrict__ ics,
                                    const float* __restrict__ vals,
                                    const float* __restrict__ weight,
                                    const float* __restrict__ bias,
                                    float*       __restrict__ out,
                                    int B)
{
    const int lane = threadIdx.x & 31;
    const int wave = threadIdx.x >> 5;
    const int b    = blockIdx.x * WAVES_PER_BLOCK + wave;
    if (b >= B) return;                       // wave-uniform exit

    // ---- lane k owns feature slot k of this batch row ----
    int   idx = ics [b * K_FEAT + lane];
    float vl  = vals[b * K_FEAT + lane];
    const bool valid = idx >= 0;              // -1 marks padding
    vl  = valid ? vl  : 0.0f;
    idx = valid ? idx : 0;
    unsigned row_byte = (unsigned)idx * (NOUT * 4u);   // gathered row byte offset

    // Warm the 32 gathered rows (1 KB each): each lane prefetches its row,
    // 4 instructions cover all lines of all 32 rows -> global_prefetch_b8.
    const char* wb = (const char*)weight;
    __builtin_prefetch(wb + row_byte,       0, 0);
    __builtin_prefetch(wb + row_byte + 256, 0, 0);
    __builtin_prefetch(wb + row_byte + 512, 0, 0);
    __builtin_prefetch(wb + row_byte + 768, 0, 0);

    const int hi = lane & 16;                 // 0 for lanes 0-15, 16 for 16-31

    // ---- B-matrix row offsets: my half-wave covers rows hi..hi+15 ----
    // voff[j] = byte offset of element (row idx[j+hi], col lane&15)
    unsigned voff[16];
    const unsigned colb = (unsigned)(lane & 15) * 4u;
    #pragma unroll
    for (int j = 0; j < 16; ++j)
        voff[j] = (unsigned)__shfl((int)row_byte, j + hi, 32) + colb;

    // ---- A-matrix: vals replicated across M (contiguous k = 16h + i) ----
    // lanes 0-15 : halves 0..15 = vals[0..15]
    // lanes 16-31: halves 0..15 = vals[16..31]
    v16h a;
    #pragma unroll
    for (int i = 0; i < 16; ++i) {
        int k = i + hi;
        a[i] = (_Float16)__shfl(vl, k, 32);
    }

    float* outb = out + (size_t)b * NOUT;

    #pragma unroll
    for (int t = 0; t < 16; ++t) {            // 16-column tiles
        // B f16 32x16: VGPR v half h, lanes0-15 -> (K=2v+h, N=lane),
        //              lanes16-31 -> (K=16+2v+h, N=lane-16)
        v16h bm;
        #pragma unroll
        for (int j = 0; j < 16; j += 2) {
            float w0 = *(const float*)(wb + voff[j]     + t * 64);
            float w1 = *(const float*)(wb + voff[j + 1] + t * 64);
            auto p = __builtin_amdgcn_cvt_pkrtz(w0, w1);  // v_cvt_pk_rtz_f16_f32
            bm[j]     = p.x;
            bm[j + 1] = p.y;
        }
        v8f c = {};   // SRC2 = inline 0
        v8f d = __builtin_amdgcn_wmma_f32_16x16x32_f16(
            /*neg_a=*/false, a, /*neg_b=*/false, bm,
            /*c_mod=*/(short)0, c, /*reuse_a=*/false, /*reuse_b=*/false);
        // D VGPR0, lanes 0-15 = (M=0, N=lane) -> out[16t + lane]
        if (lane < 16)
            outb[t * 16 + lane] = d[0] + bias[t * 16 + lane];
    }
}

extern "C" void kernel_launch(void* const* d_in, const int* in_sizes, int n_in,
                              void* d_out, int out_size, void* d_ws, size_t ws_size,
                              hipStream_t stream) {
    const int*   ics   = (const int*)  d_in[0];
    const float* vals  = (const float*)d_in[1];
    const float* wgt   = (const float*)d_in[2];
    const float* bias  = (const float*)d_in[3];
    float*       outp  = (float*)d_out;

    const int B = in_sizes[0] / K_FEAT;       // 16384
    const int blocks = (B + WAVES_PER_BLOCK - 1) / WAVES_PER_BLOCK;
    hipLaunchKernelGGL(FeatureTransformer_embbag_wmma,
                       dim3(blocks), dim3(THREADS), 0, stream,
                       ics, vals, wgt, bias, outp, B);
}
